// BitLinear_60017872994883
// MI455X (gfx1250) — matrix operand order repaired and checked by reading, not compiled, not run on hardware
//
#include <hip/hip_runtime.h>
#include <cstdint>
#include <cstddef>

// ---------------------------------------------------------------------------
// BitLinear (sign(W) * (ws @ f)) GEMM for gfx1250 / MI455X, bf16x3 WMMA.
//   x      : [256, 4096]   f32   (B*S x IN)
//   weight : [11008, 4096] f32   (OUT x IN) -> only sign() used
//   wscale : [11008, 4]    f32
//   ifac   : [4, 4096]     f32
//   out    : [256, 11008]  f32
// Memory-bound on streaming `weight` once (~180 MB @ 23.3 TB/s ~ 8 us).
// v5: fully uniform hot loop -- the 4-deep global_load_async_to_lds_b128 ring
//     always issues chunk ci+3 (tail source clamped to a valid address; that
//     ring slot is never consumed in the tail), so every iteration is a single
//     branch-free body with a constant s_wait_asynccnt 0x3. x pre-split to
//     bf16 hi/lo in d_ws in fragment-permuted order (one 32B load per A
//     fragment); single barrier per K-chunk; B fragments are two aligned
//     v8bf LDS loads each.
// f32 accuracy recovered via hi/lo bf16 split: Ah*Bh + Ah*Bl + Al*Bh.
// ---------------------------------------------------------------------------

typedef __attribute__((ext_vector_type(16))) __bf16 v16bf;
typedef __attribute__((ext_vector_type(4)))  __bf16 v4bf;
typedef __attribute__((ext_vector_type(8)))  float  v8f;

namespace {
constexpr int kM = 256;    // B*S
constexpr int kK = 4096;   // IN
constexpr int kN = 11008;  // OUT
constexpr int kR = 4;      // low rank
constexpr int BN = 32;     // output columns per workgroup
constexpr int BK = 32;     // K chunk (== WMMA K for bf16)
constexpr int PITCH = 48;  // bf16 pitch: 96 B rows -> both 32B halves aligned
constexpr int DEPTH = 4;   // async landing-pad ring depth
}

__device__ __forceinline__ void split_bf16(float v, __bf16& hi, __bf16& lo) {
  hi = (__bf16)v;
  lo = (__bf16)(v - (float)hi);
}

// Generic LDS pointer -> 32-bit DS byte address (low 32 bits of aperture form).
__device__ __forceinline__ uint32_t lds_addr(const void* p) {
  return (uint32_t)(uintptr_t)p;
}

// CDNA5 async copy: 16B per lane, global -> LDS, tracked by ASYNCcnt.
__device__ __forceinline__ void async_load_lds_b128(uint32_t lds, const float* g) {
  asm volatile("global_load_async_to_lds_b128 %0, %1, off"
               :: "v"(lds), "v"(g)
               : "memory");
}

// ---------------------------------------------------------------------------
// Pre-pass: split x (f32) into bf16 hi/lo, stored per-row in fragment-permuted
// chunk order: within each 32-K chunk, element order [0..7][16..23][8..15][24..31]
// so a lane-half's 16 fragment elements are one contiguous, 32B-aligned block.
// ---------------------------------------------------------------------------
__global__ __launch_bounds__(256)
void split_x_prepass(const float* __restrict__ x,
                     uint16_t* __restrict__ xh_,
                     uint16_t* __restrict__ xl_) {
  __bf16* xh = (__bf16*)xh_;
  __bf16* xl = (__bf16*)xl_;
  const int idx = blockIdx.x * 256 + threadIdx.x;  // one float4 per thread
  const int m  = idx >> 10;                        // kK/4 = 1024 float4 per row
  const int k4 = (idx & 1023) << 2;
  const float4 v = *(const float4*)(x + (size_t)m * kK + k4);
  // permute: swap the middle two 8-element groups of each 32-K chunk
  const int kk = k4 & 31;
  const int g  = kk >> 3;
  const int jg = (g == 1) ? 2 : ((g == 2) ? 1 : g);
  const int j0 = (k4 & ~31) + jg * 8 + (kk & 7);
  v4bf hh, ll;
  __bf16 h, l;
  split_bf16(v.x, h, l); hh[0] = h; ll[0] = l;
  split_bf16(v.y, h, l); hh[1] = h; ll[1] = l;
  split_bf16(v.z, h, l); hh[2] = h; ll[2] = l;
  split_bf16(v.w, h, l); hh[3] = h; ll[3] = l;
  *(v4bf*)(xh + (size_t)m * kK + j0) = hh;
  *(v4bf*)(xl + (size_t)m * kK + j0) = ll;
}

// ---------------------------------------------------------------------------
// Main kernel. XSPLIT=true: A fragments from pre-split bf16 arrays (one 32B
// load each). XSPLIT=false: fallback, split x in-register (if d_ws too small).
// ---------------------------------------------------------------------------
template <bool XSPLIT>
__global__ __launch_bounds__(256)
void bitlinear_bf16x3_wmma(const float* __restrict__ x,
                           const uint16_t* __restrict__ xh_,
                           const uint16_t* __restrict__ xl_,
                           const float* __restrict__ weight,
                           const float* __restrict__ wscale,
                           const float* __restrict__ ifac,
                           float* __restrict__ out) {
  __shared__ float sWraw[DEPTH][BN][BK];                   // async landing ring
  __shared__ alignas(64) __bf16 sBh[2][BN][PITCH];         // synthesized B hi
  __shared__ alignas(64) __bf16 sBl[2][BN][PITCH];         // synthesized B lo
  __shared__ float sWS[BN * kR];

  const __bf16* xh = (const __bf16*)xh_;
  const __bf16* xl = (const __bf16*)xl_;

  const int t  = threadIdx.x;
  const int n0 = blockIdx.x * BN;

  if (t < BN * kR) sWS[t] = wscale[(size_t)n0 * kR + t];
  __syncthreads();  // sWS visible (loop below has only a post-synth barrier)

  const int lane = t & 31;
  const int wave = t >> 5;        // 0..7
  const int half = lane >> 4;     // 0: lanes 0-15, 1: lanes 16-31
  const int l16  = lane & 15;
  const int m0w  = wave * 32;     // wave-private M slice

  // Cooperative staging coordinates: 8 threads per row, 16B per thread.
  // Each thread's synth reads exactly the bytes its own async copy wrote.
  const int rowS = t >> 3;        // 0..31
  const int kkS  = (t & 7) * 4;   // 0,4,...,28

  const float* wsrc = weight + (size_t)(n0 + rowS) * kK + kkS;  // + kc at use
  uint32_t ldsW[DEPTH];
#pragma unroll
  for (int d = 0; d < DEPTH; ++d) ldsW[d] = lds_addr(&sWraw[d][rowS][kkS]);

  v8f acc[2][2];
#pragma unroll
  for (int mi = 0; mi < 2; ++mi)
#pragma unroll
    for (int ni = 0; ni < 2; ++ni)
      acc[mi][ni] = (v8f){0.f, 0.f, 0.f, 0.f, 0.f, 0.f, 0.f, 0.f};

  // Prologue: async-stage chunks 0..2 (ring fill to depth-1).
  async_load_lds_b128(ldsW[0], wsrc);
  async_load_lds_b128(ldsW[1], wsrc + BK);
  async_load_lds_b128(ldsW[2], wsrc + 2 * BK);

  for (int kc = 0; kc < kK; kc += BK) {
    const int ci  = kc >> 5;
    const int buf = ci & (DEPTH - 1);

    // ---- A fragments (wave-private rows), issued early.
    v16bf ah[2], al[2];
    if constexpr (XSPLIT) {
#pragma unroll
      for (int mi = 0; mi < 2; ++mi) {
        const int row = m0w + mi * 16 + l16;
        const size_t off = (size_t)row * kK + (size_t)kc + half * 16;
        ah[mi] = *(const v16bf*)(xh + off);   // 32B aligned, one load
        al[mi] = *(const v16bf*)(xl + off);
      }
    } else {
      float4 araw[2][4];
#pragma unroll
      for (int mi = 0; mi < 2; ++mi) {
        const int row = m0w + mi * 16 + l16;
        const float* ap = x + (size_t)row * kK + (size_t)(kc + half * 8);
        araw[mi][0] = *(const float4*)(ap + 0);
        araw[mi][1] = *(const float4*)(ap + 4);
        araw[mi][2] = *(const float4*)(ap + 16);
        araw[mi][3] = *(const float4*)(ap + 20);
      }
#pragma unroll
      for (int mi = 0; mi < 2; ++mi)
#pragma unroll
        for (int q = 0; q < 4; ++q) {
          const float4 v = araw[mi][q];
          __bf16 h, l;
          split_bf16(v.x, h, l); ah[mi][q * 4 + 0] = h; al[mi][q * 4 + 0] = l;
          split_bf16(v.y, h, l); ah[mi][q * 4 + 1] = h; al[mi][q * 4 + 1] = l;
          split_bf16(v.z, h, l); ah[mi][q * 4 + 2] = h; al[mi][q * 4 + 2] = l;
          split_bf16(v.w, h, l); ah[mi][q * 4 + 3] = h; al[mi][q * 4 + 3] = l;
        }
    }

    // ---- Keep the async ring 3 chunks ahead (uniform: tail source clamped to
    //      a valid address; slot (ci+3)&3 is never consumed in the tail).
    {
      const int kpre = (kc + 3 * BK <= kK - BK) ? (kc + 3 * BK) : kc;
      async_load_lds_b128(ldsW[(ci + 3) & (DEPTH - 1)], wsrc + kpre);
      __builtin_prefetch(wsrc + kc + 8 * BK, 0, 0);  // speculative; OOB dropped
      asm volatile("s_wait_asynccnt 0x3" ::: "memory");  // in-order => chunk ci done
    }

    // ---- Synthesize B = sign(W) * (ws . f), split to bf16 hi/lo.
    {
      const float4 wv = *(const float4*)&sWraw[buf][rowS][kkS];  // own bytes
      const float4 f0 = *(const float4*)(ifac + 0 * kK + kc + kkS);
      const float4 f1 = *(const float4*)(ifac + 1 * kK + kc + kkS);
      const float4 f2 = *(const float4*)(ifac + 2 * kK + kc + kkS);
      const float4 f3 = *(const float4*)(ifac + 3 * kK + kc + kkS);
      const float s0 = sWS[rowS * kR + 0];
      const float s1 = sWS[rowS * kR + 1];
      const float s2 = sWS[rowS * kR + 2];
      const float s3 = sWS[rowS * kR + 3];
      const float fc[4][4] = {{f0.x, f0.y, f0.z, f0.w},
                              {f1.x, f1.y, f1.z, f1.w},
                              {f2.x, f2.y, f2.z, f2.w},
                              {f3.x, f3.y, f3.z, f3.w}};
      const float wc[4] = {wv.x, wv.y, wv.z, wv.w};
      v4bf hh, ll;
#pragma unroll
      for (int i = 0; i < 4; ++i) {
        const float val = s0 * fc[0][i] + s1 * fc[1][i] + s2 * fc[2][i] + s3 * fc[3][i];
        const float sv  = wc[i] > 0.f ? val : (wc[i] < 0.f ? -val : 0.f);  // jnp.sign
        __bf16 h, l;
        split_bf16(sv, h, l);
        hh[i] = h;
        ll[i] = l;
      }
      *(v4bf*)&sBh[ci & 1][rowS][kkS] = hh;
      *(v4bf*)&sBl[ci & 1][rowS][kkS] = ll;
    }
    __syncthreads();  // the only barrier per chunk: synth -> fragment reads

    // ---- B fragments: two aligned v8bf LDS loads each (k = half*8 and 16 + half*8).
    typedef __attribute__((ext_vector_type(8))) __bf16 v8bf;
    v16bf bh[2], bl[2];
#pragma unroll
    for (int ni = 0; ni < 2; ++ni) {
      const int nr = ni * 16 + l16;
      const int kb = half * 8;
      const v8bf h0 = *(const v8bf*)&sBh[ci & 1][nr][kb];
      const v8bf h1 = *(const v8bf*)&sBh[ci & 1][nr][kb + 16];
      const v8bf l0 = *(const v8bf*)&sBl[ci & 1][nr][kb];
      const v8bf l1 = *(const v8bf*)&sBl[ci & 1][nr][kb + 16];
      bh[ni] = __builtin_shufflevector(h0, h1, 0, 1, 2, 3, 4, 5, 6, 7, 8, 9, 10, 11, 12, 13, 14, 15);
      bl[ni] = __builtin_shufflevector(l0, l1, 0, 1, 2, 3, 4, 5, 6, 7, 8, 9, 10, 11, 12, 13, 14, 15);
    }

    // ---- bf16x3 WMMA: Ah*Bh + Ah*Bl + Al*Bh, f32 accumulate.
#pragma unroll
    for (int mi = 0; mi < 2; ++mi)
#pragma unroll
      for (int ni = 0; ni < 2; ++ni) {
        acc[mi][ni] = __builtin_amdgcn_wmma_f32_16x16x32_bf16(
            false, ah[mi], false, bh[ni], (short)0, acc[mi][ni], false, false);
        acc[mi][ni] = __builtin_amdgcn_wmma_f32_16x16x32_bf16(
            false, ah[mi], false, bl[ni], (short)0, acc[mi][ni], false, false);
        acc[mi][ni] = __builtin_amdgcn_wmma_f32_16x16x32_bf16(
            false, al[mi], false, bh[ni], (short)0, acc[mi][ni], false, false);
      }
  }

  // ---- epilogue: C/D layout -> element j is row (half*8 + j), col = l16.
#pragma unroll
  for (int mi = 0; mi < 2; ++mi)
#pragma unroll
    for (int ni = 0; ni < 2; ++ni) {
      const int n = n0 + ni * 16 + l16;
#pragma unroll
      for (int j = 0; j < 8; ++j) {
        const int m = m0w + mi * 16 + half * 8 + j;
        out[(size_t)m * kN + n] = acc[mi][ni][j];
      }
    }
}

extern "C" void kernel_launch(void* const* d_in, const int* in_sizes, int n_in,
                              void* d_out, int out_size, void* d_ws, size_t ws_size,
                              hipStream_t stream) {
  (void)in_sizes; (void)n_in; (void)out_size;
  const float* x      = (const float*)d_in[0];
  const float* weight = (const float*)d_in[1];
  const float* wscale = (const float*)d_in[2];
  const float* ifac   = (const float*)d_in[3];
  float* out = (float*)d_out;

  dim3 grid(kN / BN);   // 344 workgroups, one per 32 output columns
  dim3 block(256);      // 8 wave32s

  const size_t need = (size_t)2 * kM * kK * sizeof(uint16_t);  // xh + xl = 4 MB
  if (d_ws != nullptr && ws_size >= need) {
    uint16_t* xh = (uint16_t*)d_ws;
    uint16_t* xl = xh + (size_t)kM * kK;
    split_x_prepass<<<dim3((kM * kK / 4) / 256), 256, 0, stream>>>(x, xh, xl);
    bitlinear_bf16x3_wmma<true><<<grid, block, 0, stream>>>(
        x, xh, xl, weight, wscale, ifac, out);
  } else {
    bitlinear_bf16x3_wmma<false><<<grid, block, 0, stream>>>(
        x, nullptr, nullptr, weight, wscale, ifac, out);
  }
}
